// ContentEncoder_28930899706428
// MI455X (gfx1250) — matrix operand order repaired and checked by reading, not compiled, not run on hardware
//
#include <hip/hip_runtime.h>
#include <math.h>

// ---------------------------------------------------------------------------
// CDNA5 (gfx1250) fused Demucs-style encoder + VQ.
// - Encoder GEMMs on V_WMMA_F32_16X16X4_F32 (outputs are returned -> fp32).
// - VQ distance GEMM (51.5 GFLOP, argmin-only consumer) on
//   V_WMMA_F32_16X16X32_BF16 (bf16 inputs, f32 accumulate): 8x fewer matrix
//   instructions, XDL path co-executes with VALU.
// - Weights / codebook staged in LDS (padded, conflict-aware strides).
// - K-loops rolled (#pragma unroll 1) so LICM cannot hoist fragment loads.
// ---------------------------------------------------------------------------

typedef float v2f  __attribute__((ext_vector_type(2)));
typedef float v8f  __attribute__((ext_vector_type(8)));
typedef __bf16 v16bf __attribute__((ext_vector_type(16)));

__device__ __forceinline__ v8f wmma4(v2f a, v2f b, v8f c) {
  return __builtin_amdgcn_wmma_f32_16x16x4_f32(
      false, a, false, b, (short)0, c, false, false);
}

__device__ __forceinline__ float gelu_tanh(float x) {
  const float c0 = 0.7978845608028654f;  // sqrt(2/pi)
  const float c1 = 0.044715f;
  return 0.5f * x * (1.0f + tanhf(c0 * (x + c1 * x * x * x)));
}
__device__ __forceinline__ float sigm(float x) { return 1.0f / (1.0f + expf(-x)); }

// round-to-nearest-even f32 -> bf16 (as u16 in low bits)
__device__ __forceinline__ unsigned bf16r(float f) {
  unsigned u = __float_as_uint(f);
  return (u + 0x7FFFu + ((u >> 16) & 1u)) >> 16;
}
__device__ __forceinline__ unsigned pack_bf16(float lo, float hi) {
  return bf16r(lo) | (bf16r(hi) << 16);
}

// ---------------------------------------------------------------------------
// Kernel 1: henc layer 1.
// x[8,2,2048,256] --conv(8,1)s4,pad2--> [48] -> GELU -> 1x1 (96) -> GLU -> y1[8,48,512,256]
// ---------------------------------------------------------------------------
__global__ __launch_bounds__(256)
void henc1_kernel(const float* __restrict__ x, const float* __restrict__ w1,
                  const float* __restrict__ b1, const float* __restrict__ w2,
                  const float* __restrict__ b2, float* __restrict__ y1) {
  __shared__ float lA[8][48 * 16];   // per wave: A^T staging [k=ch 48][t 16]
  __shared__ float w2s[96 * 49];     // w2 padded stride 49 (conflict-free)

  const int tid  = threadIdx.x;
  const int wave = tid >> 5;
  const int lane = tid & 31;
  const int half = lane >> 4;
  const int lm   = lane & 15;

  const int b  = blockIdx.x >> 9;     // 0..7
  const int fo = blockIdx.x & 511;    // 0..511
  const float* xb = x + (size_t)b * 2 * 2048 * 256;

  for (int j = tid; j < 96 * 48; j += 256)
    w2s[(j / 48) * 49 + (j % 48)] = w2[j];
  __syncthreads();

  for (int mt = 0; mt < 2; ++mt) {
    const int tbase = (wave * 2 + mt) * 16;
    const int t     = tbase + lm;

    float xin[16];
#pragma unroll
    for (int c = 0; c < 2; ++c)
#pragma unroll
      for (int k = 0; k < 8; ++k) {
        int fi = 4 * fo - 2 + k;
        xin[c * 8 + k] = (fi >= 0 && fi < 2048)
                             ? xb[((size_t)c * 2048 + fi) * 256 + t]
                             : 0.0f;
      }
#pragma unroll 4
    for (int j = 0; j < 24; ++j) {
      int oc = half * 24 + j;
      float acc = b1[oc];
#pragma unroll
      for (int q = 0; q < 16; ++q) acc = fmaf(xin[q], w1[oc * 16 + q], acc);
      lA[wave][oc * 16 + lm] = gelu_tanh(acc);
    }
    __syncthreads();

    v8f acc[6];
#pragma unroll
    for (int nt = 0; nt < 6; ++nt) acc[nt] = {};

#pragma unroll 1
    for (int ks = 0; ks < 12; ++ks) {
      const int K0 = ks * 4 + 2 * half;
      v2f a;
      a.x = lA[wave][(K0 + 0) * 16 + lm];
      a.y = lA[wave][(K0 + 1) * 16 + lm];
#pragma unroll
      for (int nt = 0; nt < 6; ++nt) {
        const int oc = nt * 16 + lm;
        v2f bb;
        bb.x = w2s[oc * 49 + K0 + 0];
        bb.y = w2s[oc * 49 + K0 + 1];
        acc[nt] = wmma4(a, bb, acc[nt]);
      }
    }

#pragma unroll
    for (int nt = 0; nt < 3; ++nt) {
      const int ocA = nt * 16 + lm;
      const float biasA = b2[ocA];
      const float biasB = b2[ocA + 48];
      float* outp = y1 + (((size_t)b * 48 + ocA) * 512 + fo) * 256 + tbase + 8 * half;
#pragma unroll
      for (int r = 0; r < 8; ++r) {
        float za = acc[nt][r] + biasA;
        float zb = acc[nt + 3][r] + biasB;
        outp[r] = za * sigm(zb);
      }
    }
    __syncthreads();
  }
}

// ---------------------------------------------------------------------------
// Kernel 2: henc layer 2.
// ---------------------------------------------------------------------------
__global__ __launch_bounds__(256)
void henc2_kernel(const float* __restrict__ y1, const float* __restrict__ w3,
                  const float* __restrict__ b3, const float* __restrict__ w4,
                  const float* __restrict__ b4, float* __restrict__ y2) {
  __shared__ float lY[8][96 * 16];   // per wave: conv-out transpose [ch 96][t 16]
  __shared__ float w4s[192 * 97];    // w4 padded stride 97 (conflict-free)
  __shared__ float w3c[96 * 33];     // w3 K-chunk: [oc 96][K 32] pad 33

  const int tid  = threadIdx.x;
  const int wave = tid >> 5;
  const int lane = tid & 31;
  const int half = lane >> 4;
  const int lm   = lane & 15;

  const int b  = blockIdx.x >> 7;   // 0..7
  const int fo = blockIdx.x & 127;  // 0..127
  const float* y1b = y1 + (size_t)b * 48 * 512 * 256;

  for (int j = tid; j < 192 * 96; j += 256)
    w4s[(j / 96) * 97 + (j % 96)] = w4[j];

  for (int mt = 0; mt < 2; ++mt) {
    const int tbase = (wave * 2 + mt) * 16;
    const int t     = tbase + lm;

    v8f cacc[6];
#pragma unroll
    for (int nt = 0; nt < 6; ++nt) cacc[nt] = {};

#pragma unroll 1
    for (int kc = 0; kc < 12; ++kc) {       // 12 chunks x 32 K-values
      __syncthreads();
      for (int j = tid; j < 96 * 32; j += 256)
        w3c[(j >> 5) * 33 + (j & 31)] = w3[(size_t)(j >> 5) * 384 + kc * 32 + (j & 31)];
      __syncthreads();

#pragma unroll 1
      for (int ks8 = 0; ks8 < 8; ++ks8) {
        const int K0 = (kc * 8 + ks8) * 4 + 2 * half;   // global K
        const int Kl = ks8 * 4 + 2 * half;              // K within chunk
        v2f a;
        {
          int ic = K0 >> 3, k = K0 & 7;
          int fi = 4 * fo - 2 + k;
          a.x = (fi >= 0 && fi < 512)
                    ? y1b[((size_t)ic * 512 + fi) * 256 + t] : 0.0f;
          int K1 = K0 + 1;
          ic = K1 >> 3; k = K1 & 7;
          fi = 4 * fo - 2 + k;
          a.y = (fi >= 0 && fi < 512)
                    ? y1b[((size_t)ic * 512 + fi) * 256 + t] : 0.0f;
        }
#pragma unroll
        for (int nt = 0; nt < 6; ++nt) {
          const int oc = nt * 16 + lm;
          v2f bb;
          bb.x = w3c[oc * 33 + Kl + 0];
          bb.y = w3c[oc * 33 + Kl + 1];
          cacc[nt] = wmma4(a, bb, cacc[nt]);
        }
      }
    }

#pragma unroll
    for (int nt = 0; nt < 6; ++nt) {
      const int oc = nt * 16 + lm;
      const float bias = b3[oc];
#pragma unroll
      for (int r = 0; r < 8; ++r)
        lY[wave][oc * 16 + r + 8 * half] = gelu_tanh(cacc[nt][r] + bias);
    }
    __syncthreads();

#pragma unroll 1
    for (int i = 0; i < 6; ++i) {
      const int ocA = i * 16 + lm;   // value channel
      const int ocB = ocA + 96;      // gate channel
      v8f za = {}, zb = {};
#pragma unroll 1
      for (int ks = 0; ks < 24; ++ks) {
        const int K0 = ks * 4 + 2 * half;
        v2f a;
        a.x = lY[wave][(K0 + 0) * 16 + lm];
        a.y = lY[wave][(K0 + 1) * 16 + lm];
        v2f bA, bB;
        bA.x = w4s[ocA * 97 + K0 + 0];
        bA.y = w4s[ocA * 97 + K0 + 1];
        bB.x = w4s[ocB * 97 + K0 + 0];
        bB.y = w4s[ocB * 97 + K0 + 1];
        za = wmma4(a, bA, za);
        zb = wmma4(a, bB, zb);
      }
      const float biasA = b4[ocA];
      const float biasB = b4[ocB];
      float* outp = y2 + (((size_t)b * 96 + ocA) * 128 + fo) * 256 + tbase + 8 * half;
#pragma unroll
      for (int r = 0; r < 8; ++r) {
        float va = za[r] + biasA;
        float vg = zb[r] + biasB;
        outp[r] = va * sigm(vg);
      }
    }
    __syncthreads();
  }
}

// ---------------------------------------------------------------------------
// Kernel 3: codebook squared norms (1024 codes x 96 dims)
// ---------------------------------------------------------------------------
__global__ __launch_bounds__(256)
void cbnorm_kernel(const float* __restrict__ cb, float* __restrict__ cbn) {
  int k = blockIdx.x * 256 + threadIdx.x;
  if (k < 1024) {
    float s = 0.0f;
#pragma unroll 8
    for (int c = 0; c < 96; ++c) { float v = cb[k * 96 + c]; s = fmaf(v, v, s); }
    cbn[k] = s;
  }
}

// ---------------------------------------------------------------------------
// Kernel 4: VQ with bf16 WMMA (f32 accumulate).
// GEMM [262144 x 96] x [96 x 1024]: 3x v_wmma_f32_16x16x32_bf16 per 16x16 tile
// (vs 24 with fp32 WMMA). 1024-thread blocks (32 waves, 512 rows/block);
// codebook tile converted to packed bf16 at LDS-staging time.
//
// A 16x32 bf16 layout used here: lane=M; VGPR v: K = 2v+16h (contiguous
//   k order, same as B, not the ISA table order). B 32x16: lane=N; VGPR v: K = 2v+16h.
// C/D layout identical to f32 WMMA -> argmin logic unchanged.
// ---------------------------------------------------------------------------
__global__ __launch_bounds__(1024)
void vq_kernel(const float* __restrict__ y2, const float* __restrict__ cb,
               const float* __restrict__ cbn, float* __restrict__ latent,
               int* __restrict__ indices) {
  __shared__ unsigned lBh[16 * 52]; // 16 codes x 48 packed-bf16 dwords, stride 52
  __shared__ int widx[32][16];      // per-wave winning indices

  const int tid  = threadIdx.x;
  const int wave = tid >> 5;
  const int lane = tid & 31;
  const int half = lane >> 4;
  const int lm   = lane & 15;

  const int rowbase = (blockIdx.x * 32 + wave) * 16;
  const int b = rowbase >> 15;        // / 32768
  const int n = rowbase & 32767;
  const float* fb = y2 + (size_t)b * 96 * 32768 + n;

  // ---- A fragments: 3 K-steps x 8 packed dwords, bf16-converted once.
  union AFrag { unsigned u[8]; v16bf v; };
  AFrag af[3];
#pragma unroll
  for (int s = 0; s < 3; ++s) {
#pragma unroll
    for (int v = 0; v < 8; ++v) {
      const int c0 = 32 * s + 2 * v + 16 * half;
      float f0 = fb[(size_t)(c0 + 0) * 32768 + lm];
      float f1 = fb[(size_t)(c0 + 1) * 32768 + lm];
      af[s].u[v] = pack_bf16(f0, f1);
    }
  }

  float mv[8];
  int   mi[8];
#pragma unroll
  for (int r = 0; r < 8; ++r) { mv[r] = 3.4e38f; mi[r] = 0; }

#pragma unroll 1
  for (int nt = 0; nt < 64; ++nt) {
    __syncthreads();
    // stage 16 codebook rows as packed bf16: dword d = channels (2d, 2d+1)
    for (int j = tid; j < 16 * 48; j += 1024) {
      const int code = j / 48, d = j % 48;
      const float* cr = cb + (size_t)(nt * 16 + code) * 96 + 2 * d;
      lBh[code * 52 + d] = pack_bf16(cr[0], cr[1]);
    }
    __syncthreads();

    v8f acc = {};
#pragma unroll
    for (int s = 0; s < 3; ++s) {
      union { unsigned u[8]; v16bf v; } bf;
      const unsigned* p = &lBh[lm * 52 + 16 * s + 8 * half];
#pragma unroll
      for (int v = 0; v < 8; ++v) bf.u[v] = p[v];   // 2x ds_load_b128
      acc = __builtin_amdgcn_wmma_f32_16x16x32_bf16(
          false, af[s].v, false, bf.v, (short)0, acc, false, false);
    }

    const int   col = nt * 16 + lm;
    const float cn  = cbn[col];
#pragma unroll
    for (int r = 0; r < 8; ++r) {
      float s = cn - 2.0f * acc[r];   // ||v||^2 dropped (const per row)
      if (s < mv[r]) { mv[r] = s; mi[r] = col; }
    }
  }

  // Cross-lane min-reduce within each 16-lane half; tie-break to lower index.
#pragma unroll
  for (int m = 8; m >= 1; m >>= 1) {
#pragma unroll
    for (int r = 0; r < 8; ++r) {
      float ov = __shfl_xor(mv[r], m, 32);
      int   oi = __shfl_xor(mi[r], m, 32);
      if (ov < mv[r] || (ov == mv[r] && oi < mi[r])) { mv[r] = ov; mi[r] = oi; }
    }
  }

  if (lm == 0) {
#pragma unroll
    for (int r = 0; r < 8; ++r) {
      widx[wave][r + 8 * half] = mi[r];
      indices[rowbase + r + 8 * half] = mi[r];
    }
  }
  __syncthreads();

  // Gather: latent[row] = codebook[winner] (f32, exact)
#pragma unroll 1
  for (int m = 0; m < 16; ++m) {
    const int idx = widx[wave][m];
    const float* crow = cb + (size_t)idx * 96;
    float* lrow = latent + (size_t)b * 96 * 32768 + n + m;
#pragma unroll
    for (int c = lane; c < 96; c += 32)
      lrow[(size_t)c * 32768] = crow[c];
  }
}

// ---------------------------------------------------------------------------
// Launch
// ---------------------------------------------------------------------------
extern "C" void kernel_launch(void* const* d_in, const int* in_sizes, int n_in,
                              void* d_out, int out_size, void* d_ws, size_t ws_size,
                              hipStream_t stream) {
  (void)in_sizes; (void)n_in; (void)out_size; (void)ws_size;

  const float* x  = (const float*)d_in[0];
  const float* w1 = (const float*)d_in[1];
  const float* b1 = (const float*)d_in[2];
  const float* w2 = (const float*)d_in[3];
  const float* b2 = (const float*)d_in[4];
  const float* w3 = (const float*)d_in[5];
  const float* b3 = (const float*)d_in[6];
  const float* w4 = (const float*)d_in[7];
  const float* b4 = (const float*)d_in[8];
  const float* cb = (const float*)d_in[9];

  constexpr size_t NLAT = (size_t)8 * 96 * 128 * 256;  // 25,165,824
  constexpr size_t NIDX = (size_t)8 * 128 * 256;       //    262,144

  float* latent  = (float*)d_out;
  int*   indices = (int*)((float*)d_out + NLAT);
  float* beforvq = (float*)d_out + NLAT + NIDX;

  float* y1  = (float*)d_ws;                           // [8,48,512,256] = 201.3 MB
  float* cbn = y1 + (size_t)8 * 48 * 512 * 256;        // 1024 floats

  henc1_kernel<<<8 * 512, 256, 0, stream>>>(x, w1, b1, w2, b2, y1);
  henc2_kernel<<<8 * 128, 256, 0, stream>>>(y1, w3, b3, w4, b4, beforvq);
  cbnorm_kernel<<<4, 256, 0, stream>>>(cb, cbn);
  vq_kernel<<<512, 1024, 0, stream>>>(beforvq, cb, cbn, latent, indices);
}
